// SpatialGraphConv_10574209482797
// MI455X (gfx1250) — matrix operand order repaired and checked by reading, not compiled, not run on hardware
//
#include <hip/hip_runtime.h>
#include <hip/hip_bf16.h>

typedef __attribute__((ext_vector_type(16))) _Float16 v16h;
typedef __attribute__((ext_vector_type(8)))  _Float16 v8h;
typedef __attribute__((ext_vector_type(8)))  float    v8f;
typedef __attribute__((ext_vector_type(4)))  float    v4f;

#define BN 16384       // B*N
#define NN 8192        // N
#define CC 128         // C_IN == C_OUT
// exp(-d2/(2*sigma^2)) = exp2(c * d2), c = -50*log2(e), sigma = 0.1
#define CEXP (-72.13475204444817f)

// ---------------------------------------------------------------------------
// Kernel 0: pad positions [B*N][3] f32 -> [B*N] float4 (x, y, z, c*|p|^2).
// The w component pre-folds the exp2 scale so the aggregation kernel computes
// the exponent with 1 add + 3 fma per element.
// ---------------------------------------------------------------------------
__global__ void sgc_pad_pos(const float* __restrict__ pos,
                            float4* __restrict__ posp) {
    int i = blockIdx.x * blockDim.x + threadIdx.x;
    if (i < BN) {
        float px = pos[3 * i + 0];
        float py = pos[3 * i + 1];
        float pz = pos[3 * i + 2];
        float4 p;
        p.x = px; p.y = py; p.z = pz;
        p.w = CEXP * (px * px + py * py + pz * pz);
        posp[i] = p;
    }
}

// ---------------------------------------------------------------------------
// Kernel 1: h = x @ W via WMMA f16->f32, stored transposed in
// f16 so kernel 2's A-operand is one contiguous 32B load per lane:
// within each 32-wide j block of row c, j groups of 8 are stored in order
// [0..7, 8..15, 16..23, 24..31]  (group g -> slot g).
// One wave computes one 16x16 tile. Tiles: (BN/16)=1024 x (CC/16)=8.
// ---------------------------------------------------------------------------
__global__ void sgc_h_gemm(const float* __restrict__ x,   // [BN][128]
                           const float* __restrict__ W,   // [128][128]
                           _Float16* __restrict__ hT) {   // [128][BN]
    const int wave = threadIdx.x >> 5;
    const int lane = threadIdx.x & 31;
    const int tile = blockIdx.x * (blockDim.x >> 5) + wave;   // 0..8191
    const int mt = (tile >> 3) * 16;    // row tile base (bn)
    const int nt = (tile & 7) * 16;     // col tile base (c)
    const int l16 = lane & 15;
    const int hi  = lane >> 4;          // 0 or 1

    const float* xr = x + (size_t)(mt + l16) * CC;
    const int aklo = hi * 16;           // A-layout K offset per half-wave
    const int bklo = hi * 16;           // B-layout K offset per half-wave

    v8f acc = {};
    for (int kk = 0; kk < CC; kk += 32) {
        v16h a, b;
#pragma unroll
        for (int e = 0; e < 8; ++e) {
            a[e]     = (_Float16)xr[kk + aklo + e];        // K = kk+aklo+e
            a[e + 8] = (_Float16)xr[kk + aklo + 8 + e];    // K = kk+aklo+8+e
        }
#pragma unroll
        for (int e = 0; e < 16; ++e) {
            b[e] = (_Float16)W[(size_t)(kk + bklo + e) * CC + nt + l16];
        }
        acc = __builtin_amdgcn_wmma_f32_16x16x32_f16(
            false, a, false, b, (short)0, acc, false, false);
    }

    // D layout: VGPR r -> m = mt + r + hi*8, col c = nt + l16.
    // Store: 8 consecutive m's land in one 8-group of a 32-block.
    v8h hv;
#pragma unroll
    for (int r = 0; r < 8; ++r) hv[r] = (_Float16)acc[r];
    const int mstart = mt + hi * 8;
    const int block  = mstart & ~31;
    const int g      = (mstart >> 3) & 3;
    const int sg     = g;                           // identity: 0->0, 1->1, 2->2, 3->3
    *(v8h*)(hT + (size_t)(nt + l16) * BN + block + sg * 8) = hv;
}

// ---------------------------------------------------------------------------
// Kernel 2: fused Gaussian-adjacency aggregation, computed as
//   outT = hT x adjT   (D[m=c][n=i] = sum_j hT[c][j] * a[i][j])
// One wave owns a 16-row i-tile and all 128 channels (8 acc tiles) plus a
// 9th WMMA accumulator that computes the rowsum via an all-ones A operand.
// Per 32-wide j step:
//   - B operand = adjacency: lane holds fixed i = it+lane%16 and 16
//     consecutive j's; exponent = qi + qj + u.pj (1 add + 3 fma) + v_exp.
//   - A operand = hT: one contiguous v16h (32B) load per lane.
// ---------------------------------------------------------------------------
__global__ void sgc_aggregate(const float4* __restrict__ posp,  // [BN]
                              const _Float16* __restrict__ hT,  // [128][BN]
                              float* __restrict__ out) {        // [BN][128]
    const int wave = threadIdx.x >> 5;
    const int lane = threadIdx.x & 31;
    const int tile = blockIdx.x * (blockDim.x >> 5) + wave;  // 0..1023
    const int b  = tile >> 9;                // tiles per batch = N/16 = 512
    const int it = (tile & 511) * 16;        // i tile base within batch
    const int l16 = lane & 15;
    const int hi  = lane >> 4;

    const float4 pi4 = posp[b * NN + it + l16];
    const float qi = pi4.w;                  // c*|pi|^2
    const float m2c = -2.0f * CEXP;          // +100*log2(e)
    const float ux = m2c * pi4.x;
    const float uy = m2c * pi4.y;
    const float uz = m2c * pi4.z;

    const float4* pjbase = posp + b * NN;
    const _Float16* hTb  = hT + (size_t)b * NN;   // + c*BN selects channel row

    v16h ones;
#pragma unroll
    for (int e = 0; e < 16; ++e) ones[e] = (_Float16)1.0f;

    v8f acc[8] = {};   // 8 tiles of 16c x 16i -> 128 channels
    v8f accS = {};     // rowsum accumulator (every row = rowsum of col i)

    for (int jb = 0; jb < NN; jb += 32) {
        // ---- adjacency B operand: 16 exp2's per lane, consecutive j ----
        v16h bm;
        const float4* pj = pjbase + jb + hi * 16;
#pragma unroll
        for (int e = 0; e < 16; ++e) {
            float4 p = pj[e];
            float t = qi + p.w;              // c*(|pi|^2 + |pj|^2)
            t = fmaf(ux, p.x, t);
            t = fmaf(uy, p.y, t);
            t = fmaf(uz, p.z, t);            // + (-2c) * pi.pj  => c*d2
            bm[e] = (_Float16)__builtin_amdgcn_exp2f(t);
        }

        // ---- rowsum: ones x adjT ----
        accS = __builtin_amdgcn_wmma_f32_16x16x32_f16(
            false, ones, false, bm, (short)0, accS, false, false);

        // ---- 8 channel tiles: one contiguous v16h A load each ----
#pragma unroll
        for (int ct = 0; ct < 8; ++ct) {
            v16h am = *(const v16h*)(hTb + (size_t)(ct * 16 + l16) * BN
                                     + jb + hi * 16);
            acc[ct] = __builtin_amdgcn_wmma_f32_16x16x32_f16(
                false, am, false, bm, (short)0, acc[ct], false, false);
        }
    }

    // Every accumulator row of accS holds rowsum for this lane's column i.
    const float inv = 1.0f / (accS[0] + 1e-8f);

    // D layout: col n = i = it + l16 (fixed per lane), row m = c = ct*16 +
    // r + hi*8  -> 8 consecutive channels per acc tile: two float4 stores.
    float* orow = out + ((size_t)(b * NN + it + l16)) * CC;
#pragma unroll
    for (int ct = 0; ct < 8; ++ct) {
        const int c0 = ct * 16 + hi * 8;
        v4f v0 = { acc[ct][0] * inv, acc[ct][1] * inv,
                   acc[ct][2] * inv, acc[ct][3] * inv };
        v4f v1 = { acc[ct][4] * inv, acc[ct][5] * inv,
                   acc[ct][6] * inv, acc[ct][7] * inv };
        *(v4f*)(orow + c0)     = v0;
        *(v4f*)(orow + c0 + 4) = v1;
    }
}

// ---------------------------------------------------------------------------
extern "C" void kernel_launch(void* const* d_in, const int* in_sizes, int n_in,
                              void* d_out, int out_size, void* d_ws, size_t ws_size,
                              hipStream_t stream) {
    const float* x   = (const float*)d_in[0];   // [2,8192,128] f32
    const float* pos = (const float*)d_in[1];   // [2,8192,3]   f32
    const float* W   = (const float*)d_in[2];   // [128,128]    f32
    float* out = (float*)d_out;                 // [2,8192,128] f32

    // Workspace: hT (f16, 128 x 16384 = 4 MiB) then posp (float4, 256 KiB)
    _Float16* hT  = (_Float16*)d_ws;
    float4*  posp = (float4*)((char*)d_ws + (size_t)CC * BN * sizeof(_Float16));

    sgc_pad_pos<<<(BN + 255) / 256, 256, 0, stream>>>(pos, posp);
    sgc_h_gemm<<<8192 / 4, 128, 0, stream>>>(x, W, hT);
    sgc_aggregate<<<1024 / 4, 128, 0, stream>>>(posp, hT, out);
}
